// FusedCapsNet_60455959658896
// MI455X (gfx1250) — matrix operand order repaired and checked by reading, not compiled, not run on hardware
//
#include <hip/hip_runtime.h>
#include <hip/hip_bf16.h>
#include <math.h>

// ---------------------------------------------------------------------------
// CapsNet forward for gfx1250 (MI455X).
// Convs = implicit-GEMM on v_wmma_f32_16x16x32_bf16 with channel-fastest K
// order + channels-last activations so fragments load as b128 vectors.
// Routing = LDS reductions (ds_add_f32) + wave32 shuffle butterflies.
// ---------------------------------------------------------------------------

typedef __attribute__((ext_vector_type(16))) __bf16 v16bf;
typedef __attribute__((ext_vector_type(8)))  __bf16 v8bf;
typedef __attribute__((ext_vector_type(8)))  float  v8f;

union BfBits { unsigned short u; __bf16 h; };

static __device__ __forceinline__ __bf16 u2bf(unsigned short x) {
    BfBits t; t.u = x; return t.h;
}
static __device__ __forceinline__ unsigned short f2bf(float f) {
    unsigned int u = __builtin_bit_cast(unsigned int, f);
    u += 0x7fffu + ((u >> 16) & 1u);           // round-to-nearest-even
    return (unsigned short)(u >> 16);
}
static __device__ __forceinline__ float bf2f(unsigned short u) {
    unsigned int w = ((unsigned int)u) << 16;
    return __builtin_bit_cast(float, w);
}
static __device__ __forceinline__ v16bf zero16() {
    v16bf z;
#pragma unroll
    for (int h = 0; h < 16; ++h) z[h] = u2bf(0);
    return z;
}
static __device__ __forceinline__ v16bf join8(v8bf lo, v8bf hi) {
    return __builtin_shufflevector(lo, hi, 0, 1, 2, 3, 4, 5, 6, 7,
                                   8, 9, 10, 11, 12, 13, 14, 15);
}

// ---------------- workspace layout (bytes) ----------------
#define WS_H1   ((size_t)0)            // bf16 (32, 64, 64, 64c)  channels-last, 16 MB
#define WS_C2   ((size_t)16777216)     // bf16 (32, 64, 64, 128c) post-relu,     64 MB
#define WS_H2   ((size_t)83886080)     // bf16 (32, 32, 32, 128c)                 8 MB
#define WS_W2P  ((size_t)92274688)     // bf16 fragment-ready [8 ocT][18 kc][32 lane][16]
#define WS_WPP  ((size_t)92422144)     // bf16 fragment-ready [4 ocT][324 kc][32 lane][16]
#define WS_P    ((size_t)93749248)     // f32  (32, 64, 144)
#define WS_UH   ((size_t)94928896)     // f32  (32, 1152, 2, 16)

// ---------------------------------------------------------------------------
// Weight repack into fragment-ready order:
//   linear = ((ocT*NKC + kc)*32 + lane)*16 + h
//   lane L: oc = ocT*16 + (L&15);  K = kc*32 + (h&8)*2 + (L>>4)*8 + (h&7)
//   K -> ic = K & (C-1), pos = K >> log2(C), (ky,kx) = pos /% KW
// ---------------------------------------------------------------------------
__global__ void pack_w2_frag(const float* __restrict__ w2, unsigned short* __restrict__ out) {
    int idx = blockIdx.x * 256 + threadIdx.x;          // 8*18*32*16 = 73728
    int h = idx & 15;
    int lane = (idx >> 4) & 31;
    int t = idx >> 9;
    int kc = t % 18;
    int ocT = t / 18;
    int lm = lane & 15, hi = lane >> 4;
    int oc = ocT * 16 + lm;
    int K = kc * 32 + ((h & 8) << 1) + hi * 8 + (h & 7);
    int ic = K & 63;
    int pos = K >> 6;                                   // 0..8
    int ky = pos / 3, kx = pos - ky * 3;
    out[idx] = f2bf(w2[((oc * 64 + ic) * 3 + ky) * 3 + kx]);
}

__global__ void pack_wp_frag(const float* __restrict__ wp, unsigned short* __restrict__ out) {
    int idx = blockIdx.x * 256 + threadIdx.x;          // 4*324*32*16 = 663552
    int h = idx & 15;
    int lane = (idx >> 4) & 31;
    int t = idx >> 9;
    int kc = t % 324;
    int ocT = t / 324;
    int lm = lane & 15, hi = lane >> 4;
    int oc = ocT * 16 + lm;
    int K = kc * 32 + ((h & 8) << 1) + hi * 8 + (h & 7);
    int ic = K & 127;
    int pos = K >> 7;                                   // 0..80
    int ky = pos / 9, kx = pos - ky * 9;
    out[idx] = f2bf(wp[((oc * 128 + ic) * 9 + ky) * 9 + kx]);
}

// ---------------------------------------------------------------------------
// conv1 (1->64, 3x3 SAME) + ReLU + 2x2 maxpool, direct -> bf16 channels-last.
// ---------------------------------------------------------------------------
__global__ void conv1_relu_pool(const float* __restrict__ x,
                                const float* __restrict__ w1,
                                const float* __restrict__ b1,
                                unsigned short* __restrict__ h1) {
    int idx = blockIdx.x * 256 + threadIdx.x;          // 32*64*64*64
    int ch = idx & 63;                                  // channel fastest
    int xo = (idx >> 6) & 63;
    int yo = (idx >> 12) & 63;
    int bi = idx >> 18;

    float w[9];
#pragma unroll
    for (int k = 0; k < 9; ++k) w[k] = w1[ch * 9 + k];
    float bias = b1[ch];
    const float* xb = x + (size_t)bi * 128 * 128;

    float best = 0.f;                                  // max(relu) == relu(max)
#pragma unroll
    for (int dy = 0; dy < 2; ++dy) {
#pragma unroll
        for (int dx = 0; dx < 2; ++dx) {
            int Y = 2 * yo + dy, X = 2 * xo + dx;
            float acc = bias;
#pragma unroll
            for (int ky = 0; ky < 3; ++ky) {
#pragma unroll
                for (int kx = 0; kx < 3; ++kx) {
                    int iy = Y + ky - 1, ix = X + kx - 1;
                    if (iy >= 0 && iy < 128 && ix >= 0 && ix < 128)
                        acc += w[ky * 3 + kx] * xb[iy * 128 + ix];
                }
            }
            best = fmaxf(best, acc);
        }
    }
    h1[idx] = f2bf(best);
}

// ---------------------------------------------------------------------------
// conv2 (64->128, 3x3 SAME over 64x64) implicit GEMM:
//   D[spatial 4096][oc 128] = A[spatial][K=576] * B[K][oc]
// Block = 256 thr = 8 waves = 8 oc-tiles sharing one 16-wide spatial tile.
// 9 positions fully unrolled x 2 K-chunks -> 18 wmma per wave.
// Per chunk: 2x b128 activation loads + 1x 32B fragment-ready weight load.
// ---------------------------------------------------------------------------
__global__ void conv2_wmma(const unsigned short* __restrict__ h1,
                           const unsigned short* __restrict__ wb,
                           const float* __restrict__ b2,
                           unsigned short* __restrict__ c2) {
    const int tid = threadIdx.x;
    const int lane = tid & 31;
    const int ocT = tid >> 5;                 // 0..7
    const int bi = blockIdx.x >> 8;
    const int mT = blockIdx.x & 255;          // spatial tile (16 px, same row)
    const int lm = lane & 15, hi = lane >> 4;
    const int n = mT * 16 + lm;               // spatial index for A row
    const int y = n >> 6, x = n & 63;

    const unsigned short* hb = h1 + (size_t)bi * 64 * 64 * 64;
    const unsigned short* wbase = wb + ((size_t)ocT * 18) * 512 + lane * 16;

    v8f c = {};
    int kc = 0;
#pragma unroll
    for (int pos = 0; pos < 9; ++pos) {
        const int ky = pos / 3, kx = pos - (pos / 3) * 3;   // folds to constants
        const int iy = y + ky - 1, ix = x + kx - 1;
        const bool inb = (iy >= 0) & (iy < 64) & (ix >= 0) & (ix < 64);
        const unsigned short* arow = hb + (((iy * 64 + ix) << 6) + (hi << 3));
#pragma unroll
        for (int s = 0; s < 2; ++s, ++kc) {
            __builtin_prefetch(wbase + (kc + 2) * 512, 0, 1);  // global_prefetch_b8
            v16bf a;
            if (inb) {
                v8bf alo = *(const v8bf*)(arow + s * 32);       // K pairs v0..v3
                v8bf ahi = *(const v8bf*)(arow + s * 32 + 16);  // K pairs v4..v7
                a = join8(alo, ahi);
            } else {
                a = zero16();
            }
            v16bf b = *(const v16bf*)(wbase + kc * 512);        // fragment-ready
            c = __builtin_amdgcn_wmma_f32_16x16x32_bf16(false, a, false, b,
                                                        (short)0, c, false, false);
        }
    }
    // D: VGPR v -> spatial row v + 8*hi within tile; col oc = ocT*16 + lane%16.
    const int ybase = mT >> 2;
    const int xbase = (mT & 3) << 4;
    const int oc = ocT * 16 + lm;
    const float bias = b2[oc];
#pragma unroll
    for (int v = 0; v < 8; ++v) {
        int xx = xbase + v + 8 * hi;
        float val = fmaxf(c[v] + bias, 0.f);
        c2[(((size_t)bi * 64 + ybase) * 64 + xx) * 128 + oc] = f2bf(val);
    }
}

// ---------------------------------------------------------------------------
// 2x2 maxpool, channels-last bf16 (inputs >= 0 post-relu).
// ---------------------------------------------------------------------------
__global__ void maxpool2_bf16(const unsigned short* __restrict__ c2,
                              unsigned short* __restrict__ h2) {
    int idx = blockIdx.x * 256 + threadIdx.x;          // 32*32*32*128
    int ch = idx & 127;
    int xo = (idx >> 7) & 31;
    int yo = (idx >> 12) & 31;
    int bi = idx >> 17;
    const unsigned short* src = c2 + (size_t)bi * 64 * 64 * 128;
    float m = 0.f;
#pragma unroll
    for (int dy = 0; dy < 2; ++dy)
#pragma unroll
        for (int dx = 0; dx < 2; ++dx)
            m = fmaxf(m, bf2f(src[(((2 * yo + dy) * 64) + (2 * xo + dx)) * 128 + ch]));
    h2[idx] = f2bf(m);
}

// ---------------------------------------------------------------------------
// Primary caps: 9x9 stride-2 VALID conv (128->64) over 32x32 -> (64,12,12) f32.
//   D[spatial 144][oc 64] = A[sp][K=10368] * B[K][oc];  81 positions x 4 chunks.
// Block = 128 thr = 4 waves = 4 oc-tiles sharing one spatial tile.
// ---------------------------------------------------------------------------
__global__ void pcaps_wmma(const unsigned short* __restrict__ h2,
                           const unsigned short* __restrict__ wb,
                           const float* __restrict__ bp,
                           float* __restrict__ p) {
    const int tid = threadIdx.x;
    const int lane = tid & 31;
    const int ocT = tid >> 5;                 // 0..3
    const int bi = blockIdx.x / 9;
    const int mT = blockIdx.x - bi * 9;       // spatial tile 0..8
    const int lm = lane & 15, hi = lane >> 4;
    const int n = mT * 16 + lm;               // spatial 0..143
    const int y = n / 12, x = n - y * 12;

    const unsigned short* hb = h2 + (size_t)bi * 32 * 32 * 128;
    const unsigned short* wbase = wb + ((size_t)ocT * 324) * 512 + lane * 16;

    v8f c = {};
    int kc = 0;
    for (int pos = 0; pos < 81; ++pos) {
        int ky = pos / 9, kx = pos - (pos / 9) * 9;
        int Y = 2 * y + ky, X = 2 * x + kx;            // always in [0,31] (VALID)
        const unsigned short* arow = hb + (((Y * 32 + X) << 7) + (hi << 3));
#pragma unroll
        for (int s = 0; s < 4; ++s, ++kc) {
            __builtin_prefetch(wbase + (kc + 4) * 512, 0, 1);
            v8bf alo = *(const v8bf*)(arow + s * 32);
            v8bf ahi = *(const v8bf*)(arow + s * 32 + 16);
            v16bf a = join8(alo, ahi);
            v16bf b = *(const v16bf*)(wbase + kc * 512);
            c = __builtin_amdgcn_wmma_f32_16x16x32_bf16(false, a, false, b,
                                                        (short)0, c, false, false);
        }
    }
    const int oc = ocT * 16 + lm;
    const float bias = bp[oc];
#pragma unroll
    for (int v = 0; v < 8; ++v) {
        int sp = mT * 16 + v + 8 * hi;                 // 0..143
        p[((size_t)bi * 64 + oc) * 144 + sp] = c[v] + bias;
    }
}

// ---------------------------------------------------------------------------
// Squash -> u_hat -> 3 rounds of dynamic routing -> log_softmax. 1 block/batch.
// ---------------------------------------------------------------------------
__global__ void caps_route(const float* __restrict__ p,
                           const float* __restrict__ Wc,
                           float* __restrict__ uh_g,
                           float* __restrict__ out) {
    __shared__ float bij[1152 * 2];
    __shared__ float s_s[32];
    __shared__ float v_s[32];

    const int tid = threadIdx.x;
    const int bi = blockIdx.x;
    float* uh = uh_g + (size_t)bi * 1152 * 32;
    const float* pb = p + (size_t)bi * 64 * 144;

    // u = squash(p reshaped); u_hat[i][j][d] = sum_e Wc[i][j][d][e] * u[i][e]
    for (int i = tid; i < 1152; i += 256) {
        int s = i >> 3, q = i & 7;                 // spatial, cap
        float uv[8];
        float sn = 0.f;
#pragma unroll
        for (int e = 0; e < 8; ++e) {
            uv[e] = pb[(q * 8 + e) * 144 + s];
            sn += uv[e] * uv[e];
        }
        float scale = sn / ((1.f + sn) * sqrtf(sn + 1e-8f));
#pragma unroll
        for (int e = 0; e < 8; ++e) uv[e] *= scale;
        const float* wrow = Wc + (size_t)i * 2 * 16 * 8;
#pragma unroll
        for (int j = 0; j < 2; ++j)
            for (int d = 0; d < 16; ++d) {
                float acc = 0.f;
#pragma unroll
                for (int e = 0; e < 8; ++e)
                    acc += wrow[(j * 16 + d) * 8 + e] * uv[e];
                uh[(i * 2 + j) * 16 + d] = acc;
            }
        bij[2 * i] = 0.f;
        bij[2 * i + 1] = 0.f;
    }
    __syncthreads();

    for (int r = 0; r < 3; ++r) {
        if (tid < 32) s_s[tid] = 0.f;
        __syncthreads();

        float acc[32];
#pragma unroll
        for (int t = 0; t < 32; ++t) acc[t] = 0.f;

        for (int i = tid; i < 1152; i += 256) {
            float b0 = bij[2 * i], b1 = bij[2 * i + 1];
            float mx = fmaxf(b0, b1);
            float e0 = expf(b0 - mx), e1 = expf(b1 - mx);
            float inv = 1.f / (e0 + e1);
            float c0 = e0 * inv, c1 = e1 * inv;
            const float* u0 = &uh[(size_t)i * 32];
#pragma unroll
            for (int d = 0; d < 16; ++d) {
                acc[d]      += c0 * u0[d];
                acc[16 + d] += c1 * u0[16 + d];
            }
        }
        // wave32 butterfly reduction of the 32 accumulators
#pragma unroll
        for (int t = 0; t < 32; ++t) {
#pragma unroll
            for (int off = 16; off > 0; off >>= 1)
                acc[t] += __shfl_xor(acc[t], off, 32);
        }
        if ((tid & 31) == 0) {
#pragma unroll
            for (int t = 0; t < 32; ++t) atomicAdd(&s_s[t], acc[t]);
        }
        __syncthreads();

        if (tid < 2) {
            int j = tid;
            float sn = 0.f;
#pragma unroll
            for (int d = 0; d < 16; ++d) sn += s_s[j * 16 + d] * s_s[j * 16 + d];
            float scale = sn / ((1.f + sn) * sqrtf(sn + 1e-8f));
#pragma unroll
            for (int d = 0; d < 16; ++d) v_s[j * 16 + d] = s_s[j * 16 + d] * scale;
        }
        __syncthreads();

        if (r < 2) {
            for (int i = tid; i < 1152; i += 256) {
                const float* u0 = &uh[(size_t)i * 32];
                float d0 = 0.f, d1 = 0.f;
#pragma unroll
                for (int d = 0; d < 16; ++d) {
                    d0 += u0[d] * v_s[d];
                    d1 += u0[16 + d] * v_s[16 + d];
                }
                bij[2 * i] += d0;
                bij[2 * i + 1] += d1;
            }
            __syncthreads();
        }
    }

    if (tid == 0) {
        float q0 = 0.f, q1 = 0.f;
#pragma unroll
        for (int d = 0; d < 16; ++d) {
            q0 += v_s[d] * v_s[d];
            q1 += v_s[16 + d] * v_s[16 + d];
        }
        float l0 = sqrtf(q0), l1 = sqrtf(q1);
        float mx = fmaxf(l0, l1);
        float lse = logf(expf(l0 - mx) + expf(l1 - mx));
        out[bi * 2 + 0] = (l0 - mx) - lse;
        out[bi * 2 + 1] = (l1 - mx) - lse;
    }
}

// ---------------------------------------------------------------------------
extern "C" void kernel_launch(void* const* d_in, const int* in_sizes, int n_in,
                              void* d_out, int out_size, void* d_ws, size_t ws_size,
                              hipStream_t stream) {
    (void)in_sizes; (void)n_in; (void)out_size; (void)ws_size;
    const float* x  = (const float*)d_in[0];
    const float* w1 = (const float*)d_in[1];
    const float* b1 = (const float*)d_in[2];
    const float* w2 = (const float*)d_in[3];
    const float* b2 = (const float*)d_in[4];
    const float* wp = (const float*)d_in[5];
    const float* bp = (const float*)d_in[6];
    const float* Wc = (const float*)d_in[7];
    float* out = (float*)d_out;

    char* ws = (char*)d_ws;
    unsigned short* h1   = (unsigned short*)(ws + WS_H1);
    unsigned short* c2   = (unsigned short*)(ws + WS_C2);
    unsigned short* h2   = (unsigned short*)(ws + WS_H2);
    unsigned short* w2pk = (unsigned short*)(ws + WS_W2P);
    unsigned short* wppk = (unsigned short*)(ws + WS_WPP);
    float* p  = (float*)(ws + WS_P);
    float* uh = (float*)(ws + WS_UH);

    pack_w2_frag<<<288, 256, 0, stream>>>(w2, w2pk);
    pack_wp_frag<<<2592, 256, 0, stream>>>(wp, wppk);
    conv1_relu_pool<<<32768, 256, 0, stream>>>(x, w1, b1, h1);
    conv2_wmma<<<8192, 256, 0, stream>>>(h1, w2pk, b2, c2);     // 65536 waves
    maxpool2_bf16<<<16384, 256, 0, stream>>>(c2, h2);
    pcaps_wmma<<<288, 128, 0, stream>>>(h2, wppk, bp, p);       // 1152 waves
    caps_route<<<32, 256, 0, stream>>>(p, Wc, uh, out);
}
